// BLHmmLm_84662395339065
// MI455X (gfx1250) — matrix operand order repaired and checked by reading, not compiled, not run on hardware
//
#include <hip/hip_runtime.h>
#include <hip/hip_bf16.h>

// ---------------------------------------------------------------------------
// Problem constants: C=4096, V=32000, H=256, D=128, N=16, T=256
// ---------------------------------------------------------------------------
#define CC 4096
#define VV 32000
#define HH 256
#define DD 128
#define NN 16
#define TT 256

typedef _Float16 h16;
typedef __attribute__((ext_vector_type(16))) _Float16 v16h;
typedef __attribute__((ext_vector_type(8)))  float    v8f;
typedef unsigned int u32x4 __attribute__((ext_vector_type(4)));
typedef int          i32x8 __attribute__((ext_vector_type(8)));
typedef int          i32x4 __attribute__((ext_vector_type(4)));

// ---------------------------------------------------------------------------
// WMMA fragment helpers for V_WMMA_F32_16X16X32_F16 (wave32).
// A (16x32 f16): lanes 0-15 M=lane; VGPR0-7 = K[kb+hi*16..+16),
// hi = lane>>4.  B^T rows are contiguous 16-half chunks.
// C/D (16x16 f32): lane<16 -> N=lane, rows M = hi*8 + r.
// ---------------------------------------------------------------------------
__device__ __forceinline__ v16h frag_a(const h16* __restrict__ row, int kb, int hi) {
  v16h v;
#pragma unroll
  for (int i = 0; i < 16; ++i) v[i] = row[kb + hi * 16 + i];
  return v;
}

__device__ __forceinline__ v16h frag_bt(const h16* __restrict__ trow, int kb, int hi) {
  return *(const v16h*)(trow + kb + hi * 16);  // 32B-aligned contiguous
}

__device__ __forceinline__ v8f wmma16(v16h a, v16h b, v8f c) {
  return __builtin_amdgcn_wmma_f32_16x16x32_f16(false, a, false, b, (short)0, c,
                                                false, false);
}

// ---------------------------------------------------------------------------
// Tensor Data Mover: 1 contiguous tile of n_h16 fp16 elements -> LDS.
// D# per CDNA5 ISA ch.8: group0 {count=1, lds_addr, global_addr, type=2},
// group1 {data_size=2B, tensor_dim0=n, tensor_dim1=1, tile_dim0=n, tile_dim1=1,
//         tensor_dim0_stride=n}; groups 2/3 zero (<=2D tensor).
// ---------------------------------------------------------------------------
__device__ __forceinline__ void tdm_load_h16(unsigned lds_addr, const void* gptr,
                                             unsigned n_h16) {
  unsigned long long ga = (unsigned long long)gptr;
  u32x4 g0;
  g0[0] = 1u;                                           // count=1, user mode
  g0[1] = lds_addr;                                     // LDS byte address
  g0[2] = (unsigned)(ga & 0xffffffffu);                 // global_addr[31:0]
  g0[3] = (unsigned)((ga >> 32) & 0x1ffffffu) | (2u << 30);  // addr[56:32]|type=2
  i32x8 g1;
  g1[0] = 0x00010000;                    // wg_mask=0, data_size=1 (2 bytes)
  g1[1] = (int)((n_h16 & 0xffffu) << 16);        // tensor_dim0[15:0]
  g1[2] = (int)((n_h16 >> 16) & 0xffffu) | (1 << 16);  // t_dim0[31:16] | t_dim1=1
  g1[3] = (int)(n_h16 << 16);                    // t_dim1[31:16]=0 | tile_dim0=n
  g1[4] = 1;                                     // tile_dim1=1, tile_dim2=0
  g1[5] = (int)n_h16;                            // tensor_dim0_stride[31:0]
  g1[6] = 0;
  g1[7] = 0;
  i32x4 z4 = {0, 0, 0, 0};
#if __clang_major__ >= 23
  i32x8 z8 = {0, 0, 0, 0, 0, 0, 0, 0};
  __builtin_amdgcn_tensor_load_to_lds(g0, g1, z4, z4, z8, 0);
#else
  __builtin_amdgcn_tensor_load_to_lds(g0, g1, z4, z4, 0);
#endif
}

__device__ __forceinline__ unsigned lds_off(const void* p) {
  return (unsigned)(unsigned long long)p;  // low 32 bits = LDS byte offset
}

// ---------------------------------------------------------------------------
// Prep: WT[n*K + k] = (h16) W[k*N + n]   (transpose + f16 convert, one-time)
// ---------------------------------------------------------------------------
__global__ void k_convT(const float* __restrict__ W, h16* __restrict__ WT,
                        int K, int Ncols) {
  int idx = blockIdx.x * 256 + threadIdx.x;
  if (idx < K * Ncols) {
    int n = idx / K, k = idx - n * K;
    WT[idx] = (h16)W[k * Ncols + n];
  }
}

// ---------------------------------------------------------------------------
// k_linear: OUT[M,256] = mode( IN[M,256] @ W + b [, +RES] ), W given as
// f16-transposed WT[256][256]. mode 0: bias; 1: relu; 2: relu + RES.
// grid (ceil(M/16), 2), block 256: 8 waves x one 16x16 tile of 128 cols.
// ---------------------------------------------------------------------------
__global__ void __launch_bounds__(256) k_linear(
    const float* __restrict__ IN, const h16* __restrict__ WT,
    const float* __restrict__ B, const float* __restrict__ RES,
    float* __restrict__ OUT, int M, int mode) {
  __shared__ h16 Ain[16][HH];
  const int tid = threadIdx.x;
  const int lane = tid & 31, wv = tid >> 5;
  const int hi = lane >> 4, l15 = lane & 15;
  const int r0 = blockIdx.x * 16;
  const int c0 = blockIdx.y * 128;

  for (int idx = tid; idx < 16 * HH; idx += 256) {
    int r = idx >> 8, c = idx & 255;
    float v = (r0 + r < M) ? IN[(r0 + r) * HH + c] : 0.f;
    Ain[r][c] = (h16)v;
  }
  __syncthreads();

  const int col = c0 + wv * 16 + l15;
  const h16* wtcol = WT + (size_t)col * HH;
  v8f acc = {0.f, 0.f, 0.f, 0.f, 0.f, 0.f, 0.f, 0.f};
#pragma unroll
  for (int kb = 0; kb < HH; kb += 32)
    acc = wmma16(frag_a(&Ain[l15][0], kb, hi), frag_bt(wtcol, kb, hi), acc);

  const float bb = B[col];
#pragma unroll
  for (int r = 0; r < 8; ++r) {
    int row = r0 + hi * 8 + r;
    if (row < M) {
      float v = acc[r] + bb;
      if (mode >= 1) v = v > 0.f ? v : 0.f;
      if (mode == 2) v += RES[row * HH + col];
      OUT[row * HH + col] = v;
    }
  }
}

// ---------------------------------------------------------------------------
// k_norm_project: OUT[M,128] = (IN[M,256]/||row||) @ proj - 0.5, proj given as
// f16-transposed PT[128][256].
// ---------------------------------------------------------------------------
__global__ void __launch_bounds__(256) k_norm_project(
    const float* __restrict__ IN, const h16* __restrict__ PT,
    float* __restrict__ OUT, int M) {
  __shared__ h16 Ain[16][HH];
  __shared__ float nrm[16][16];
  __shared__ float rn[16];
  const int tid = threadIdx.x;
  const int lane = tid & 31, wv = tid >> 5;
  const int hi = lane >> 4, l15 = lane & 15;
  const int r0 = blockIdx.x * 16;

  {
    int r = tid >> 4, cb = tid & 15;
    float s = 0.f;
    for (int c = cb; c < HH; c += 16) {
      float v = (r0 + r < M) ? IN[(r0 + r) * HH + c] : 0.f;
      Ain[r][c] = (h16)v;
      s += v * v;
    }
    nrm[r][cb] = s;
  }
  __syncthreads();
  if (tid < 16) {
    float s = 0.f;
#pragma unroll
    for (int j = 0; j < 16; ++j) s += nrm[tid][j];
    rn[tid] = (s > 0.f) ? rsqrtf(s) : 0.f;
  }
  __syncthreads();

  const int col = wv * 16 + l15;
  const h16* ptcol = PT + (size_t)col * HH;
  v8f acc = {0.f, 0.f, 0.f, 0.f, 0.f, 0.f, 0.f, 0.f};
#pragma unroll
  for (int kb = 0; kb < HH; kb += 32)
    acc = wmma16(frag_a(&Ain[l15][0], kb, hi), frag_bt(ptcol, kb, hi), acc);
#pragma unroll
  for (int r = 0; r < 8; ++r) {
    int row = r0 + hi * 8 + r;
    if (row < M) OUT[row * DD + col] = acc[r] * rn[hi * 8 + r] - 0.5f;
  }
}

// ---------------------------------------------------------------------------
// k_exp_colsum: Wout[i,d] = (h16) exp(L[i,d]);  S[d] += column sums.
// ---------------------------------------------------------------------------
__global__ void __launch_bounds__(256) k_exp_colsum(
    const float* __restrict__ L, h16* __restrict__ Wout,
    float* __restrict__ S, int M) {
  __shared__ float cs[DD];
  const int tid = threadIdx.x;
  if (tid < DD) cs[tid] = 0.f;
  __syncthreads();
  const long total = (long)M * DD;
  const long base = (long)blockIdx.x * 4096;
#pragma unroll
  for (int k = 0; k < 16; ++k) {
    long idx = base + (long)k * 256 + tid;
    if (idx < total) {
      float v = __expf(L[idx]);
      Wout[idx] = (h16)v;
      atomicAdd(&cs[(int)(idx & (DD - 1))], v);
    }
  }
  __syncthreads();
  if (tid < DD) atomicAdd(&S[tid], cs[tid]);
}

// ---------------------------------------------------------------------------
// k_make_u: U[r,d] = exp(L[r,d]-mx) / ( sum_d exp(L[r,d]-mx)*S[d] )  -> f16
// One wave per row, 4 elems/lane (wave32 shuffles).
// ---------------------------------------------------------------------------
__global__ void __launch_bounds__(256) k_make_u(
    const float* __restrict__ L, const float* __restrict__ S,
    h16* __restrict__ U, int M) {
  const int tid = threadIdx.x, lane = tid & 31, wv = tid >> 5;
  const int row = blockIdx.x * 8 + wv;
  if (row >= M) return;
  float l[4];
  float mx = -1e30f;
#pragma unroll
  for (int j = 0; j < 4; ++j) {
    l[j] = L[row * DD + lane + 32 * j];
    mx = fmaxf(mx, l[j]);
  }
#pragma unroll
  for (int o = 16; o >= 1; o >>= 1) mx = fmaxf(mx, __shfl_xor(mx, o, 32));
  float den = 0.f;
#pragma unroll
  for (int j = 0; j < 4; ++j) {
    l[j] = __expf(l[j] - mx);
    den += l[j] * S[lane + 32 * j];
  }
#pragma unroll
  for (int o = 16; o >= 1; o >>= 1) den += __shfl_xor(den, o, 32);
  const float inv = 1.f / den;
#pragma unroll
  for (int j = 0; j < 4; ++j) U[row * DD + lane + 32 * j] = (h16)(l[j] * inv);
}

// ---------------------------------------------------------------------------
__global__ void k_zero(float* __restrict__ p, long n) {
  long i = (long)blockIdx.x * blockDim.x + threadIdx.x;
  if (i < n) p[i] = 0.f;
}

// h2 <- Ux replicated (t=0 state); h0 <- 0 (t=0 write buffer); Zi <- 1.
__global__ void __launch_bounds__(256) k_init(
    const h16* __restrict__ Ux, float* __restrict__ h2,
    float* __restrict__ h0, float* __restrict__ Zi) {
  const int tid = threadIdx.x;
  for (int i = tid; i < NN * DD; i += 256) {
    h2[i] = (float)Ux[i & (DD - 1)];
    h0[i] = 0.f;
  }
  if (tid < NN) Zi[tid] = 1.f;
}

// ---------------------------------------------------------------------------
// k_step: one factorized HMM forward step.
//  S1[n,c] = (hprev[n,:]/Zprev[n]) · Wn[c,:]    (transition numerator)
//  S2[n,c] = Wt[tok[n],:] · Ue[c,:]             (emission prob at token)
//  w = S1*S2;  Znext[n] += sum_c w;  hnext[n,:] += sum_c w[n,c]*Us[c,:]
// 64 blocks x 64 states. Wn/Ue/Us tiles (16KB each, contiguous) staged by TDM.
// ---------------------------------------------------------------------------
__global__ void __launch_bounds__(256) k_step(
    int t, const h16* __restrict__ Us, const h16* __restrict__ Wn,
    const h16* __restrict__ Ue, const h16* __restrict__ Wt,
    const int* __restrict__ text, const float* __restrict__ hprev,
    float* __restrict__ hnext, float* __restrict__ hzero,
    const float* __restrict__ Zprev, float* __restrict__ Znext) {
  __shared__ h16 hn[NN][DD];
  __shared__ h16 wt[NN][DD];
  __shared__ h16 WnB[64][DD];
  __shared__ h16 UeB[64][DD];
  __shared__ h16 UsB[64][DD];
  __shared__ float S1f[NN][64];
  __shared__ float S2f[NN][64];
  __shared__ h16 Wh[NN][64];
  __shared__ float zp[NN];
  __shared__ int tk[NN];
  __shared__ float zrow[NN];

  const int tid = threadIdx.x, lane = tid & 31, wv = tid >> 5;
  const int hi = lane >> 4, l15 = lane & 15;
  const int cbase = blockIdx.x * 64;

  // Kick off the three 16KB tile DMAs first (wave 0 only; TDM ignores EXEC).
  if (wv == 0) {
    tdm_load_h16(lds_off(&WnB[0][0]), Wn + (size_t)cbase * DD, 64 * DD);
    tdm_load_h16(lds_off(&UeB[0][0]), Ue + (size_t)cbase * DD, 64 * DD);
    tdm_load_h16(lds_off(&UsB[0][0]), Us + (size_t)cbase * DD, 64 * DD);
  }
  if (blockIdx.x == 0) {  // zero the buffer step t+1 accumulates into
    for (int i = tid; i < NN * DD; i += 256) hzero[i] = 0.f;
  }
  if (tid < NN) {
    zp[tid] = Zprev[tid];
    tk[tid] = text[tid * TT + t];
    zrow[tid] = 0.f;
  }
  __syncthreads();

  for (int i = tid; i < NN * DD; i += 256) {
    int n = i >> 7, d = i & (DD - 1);
    hn[n][d] = (h16)(hprev[i] / zp[n]);
    wt[n][d] = Wt[(size_t)tk[n] * DD + d];
  }
  if (wv == 0) __builtin_amdgcn_s_wait_tensorcnt(0);
  __syncthreads();

  // GEMM1: waves 0-3 -> S1 = hn @ WnB^T ; waves 4-7 -> S2 = wt @ UeB^T
  {
    const int j = (wv & 3) * 16 + l15;
    const h16* arow = (wv < 4) ? &hn[l15][0] : &wt[l15][0];
    const h16* brow = (wv < 4) ? &WnB[j][0] : &UeB[j][0];
    v8f acc = {0.f, 0.f, 0.f, 0.f, 0.f, 0.f, 0.f, 0.f};
#pragma unroll
    for (int kb = 0; kb < DD; kb += 32)
      acc = wmma16(frag_a(arow, kb, hi), frag_bt(brow, kb, hi), acc);
    float* dst = (wv < 4) ? &S1f[0][0] : &S2f[0][0];
#pragma unroll
    for (int r = 0; r < 8; ++r) dst[(hi * 8 + r) * 64 + j] = acc[r];
  }
  __syncthreads();

  for (int i = tid; i < NN * 64; i += 256) {
    int n = i >> 6, c = i & 63;
    float w = S1f[n][c] * S2f[n][c];
    Wh[n][c] = (h16)w;
    atomicAdd(&zrow[n], w);
  }
  __syncthreads();
  if (tid < NN) atomicAdd(&Znext[tid], zrow[tid]);

  // GEMM2: ACC[16,128] = Wh[16,64] @ UsB[64,128]; wave w -> cols [16w,16w+16)
  {
    const int col = wv * 16 + l15;
    v8f acc = {0.f, 0.f, 0.f, 0.f, 0.f, 0.f, 0.f, 0.f};
#pragma unroll
    for (int kb = 0; kb < 64; kb += 32) {
      v16h a = frag_a(&Wh[l15][0], kb, hi);
      v16h b;
#pragma unroll
      for (int i = 0; i < 16; ++i) b[i] = UsB[kb + hi * 16 + i][col];
      acc = wmma16(a, b, acc);
    }
#pragma unroll
    for (int r = 0; r < 8; ++r)
      atomicAdd(&hnext[(hi * 8 + r) * DD + col], acc[r]);
  }
}

// ---------------------------------------------------------------------------
__global__ void __launch_bounds__(256) k_evidence(
    const float* __restrict__ Z, const unsigned char* __restrict__ mask,
    float* __restrict__ out) {
  __shared__ float red[256];
  const int tid = threadIdx.x;
  float s = 0.f;
  for (int i = tid; i < TT * NN; i += 256) {
    int t = i >> 4, n = i & 15;
    if (mask[n * TT + t]) s += __logf(Z[i]);
  }
  red[tid] = s;
  __syncthreads();
  for (int o = 128; o >= 1; o >>= 1) {
    if (tid < o) red[tid] += red[tid + o];
    __syncthreads();
  }
  if (tid == 0) out[0] = red[0];
}

// ---------------------------------------------------------------------------
extern "C" void kernel_launch(void* const* d_in, const int* in_sizes, int n_in,
                              void* d_out, int out_size, void* d_ws,
                              size_t ws_size, hipStream_t stream) {
  (void)in_sizes; (void)n_in; (void)out_size; (void)ws_size;
  const int* text = (const int*)d_in[0];
  const unsigned char* mask = (const unsigned char*)d_in[1];
  const float* start_emb = (const float*)d_in[2];
  const float* sw[5] = {(const float*)d_in[3], (const float*)d_in[5],
                        (const float*)d_in[7], (const float*)d_in[9],
                        (const float*)d_in[11]};
  const float* sb[5] = {(const float*)d_in[4], (const float*)d_in[6],
                        (const float*)d_in[8], (const float*)d_in[10],
                        (const float*)d_in[12]};
  const float* tw[4] = {(const float*)d_in[13], (const float*)d_in[15],
                        (const float*)d_in[17], (const float*)d_in[19]};
  const float* tb[4] = {(const float*)d_in[14], (const float*)d_in[16],
                        (const float*)d_in[18], (const float*)d_in[20]};
  const float* state_emb = (const float*)d_in[21];
  const float* next_state_emb = (const float*)d_in[22];
  const float* pret = (const float*)d_in[23];
  const float* term = (const float*)d_in[24];
  const float* proj = (const float*)d_in[25];

  char* base = (char*)d_ws;
  size_t off = 0;
  auto alloc = [&](size_t bytes) {
    void* p = base + off;
    off = (off + bytes + 255) & ~(size_t)255;
    return p;
  };
  float* x0   = (float*)alloc(HH * 4);
  float* tmp1 = (float*)alloc(HH * 4);
  float* x1   = (float*)alloc(HH * 4);
  float* tmp2 = (float*)alloc(HH * 4);
  float* fxs  = (float*)alloc(HH * 4);
  float* bufA = (float*)alloc((size_t)CC * HH * 4);
  float* bufB = (float*)alloc((size_t)CC * HH * 4);
  float* bufC = (float*)alloc((size_t)CC * HH * 4);
  float* Lxs  = (float*)alloc((size_t)CC * DD * 4);
  float* Lyn  = (float*)alloc((size_t)CC * DD * 4);
  float* Lxe  = (float*)alloc((size_t)CC * DD * 4);
  float* Lyt  = (float*)alloc((size_t)VV * DD * 4);
  float* Lx0  = (float*)alloc(DD * 4);
  h16* swT[5]; for (int i = 0; i < 5; ++i) swT[i] = (h16*)alloc((size_t)HH * HH * 2);
  h16* twT[4]; for (int i = 0; i < 4; ++i) twT[i] = (h16*)alloc((size_t)HH * HH * 2);
  h16* projT = (h16*)alloc((size_t)DD * HH * 2);
  h16* Ush = (h16*)alloc((size_t)CC * DD * 2);
  h16* Wnh = (h16*)alloc((size_t)CC * DD * 2);
  h16* Ueh = (h16*)alloc((size_t)CC * DD * 2);
  h16* Wth = (h16*)alloc((size_t)VV * DD * 2);
  h16* Uxh = (h16*)alloc(DD * 2);
  float* Sn = (float*)alloc(DD * 4);
  float* St = (float*)alloc(DD * 4);
  float* Zinit = (float*)alloc(NN * 4);
  float* Zstep = (float*)alloc((size_t)TT * NN * 4);
  float* hb = (float*)alloc((size_t)3 * NN * DD * 4);

  const dim3 blk(256);
  const int wgrid = (HH * HH + 255) / 256;

  // One-time weight transpose + f16 conversion
  for (int i = 0; i < 5; ++i)
    k_convT<<<wgrid, blk, 0, stream>>>(sw[i], swT[i], HH, HH);
  for (int i = 0; i < 4; ++i)
    k_convT<<<wgrid, blk, 0, stream>>>(tw[i], twT[i], HH, HH);
  k_convT<<<(HH * DD + 255) / 256, blk, 0, stream>>>(proj, projT, HH, DD);

  // start(): Linear + ResLayer + ResLayer (M=1)
  k_linear<<<dim3(1, 2), blk, 0, stream>>>(start_emb, swT[0], sb[0], nullptr, x0, 1, 0);
  k_linear<<<dim3(1, 2), blk, 0, stream>>>(x0, swT[1], sb[1], nullptr, tmp1, 1, 1);
  k_linear<<<dim3(1, 2), blk, 0, stream>>>(tmp1, swT[2], sb[2], x0, x1, 1, 2);
  k_linear<<<dim3(1, 2), blk, 0, stream>>>(x1, swT[3], sb[3], nullptr, tmp2, 1, 1);
  k_linear<<<dim3(1, 2), blk, 0, stream>>>(tmp2, swT[4], sb[4], x1, fxs, 1, 2);

  // terminal_mlp: 2 ResLayers on preterminal_emb [C,H]
  k_linear<<<dim3(CC / 16, 2), blk, 0, stream>>>(pret, twT[0], tb[0], nullptr, bufA, CC, 1);
  k_linear<<<dim3(CC / 16, 2), blk, 0, stream>>>(bufA, twT[1], tb[1], pret, bufB, CC, 2);
  k_linear<<<dim3(CC / 16, 2), blk, 0, stream>>>(bufB, twT[2], tb[2], nullptr, bufA, CC, 1);
  k_linear<<<dim3(CC / 16, 2), blk, 0, stream>>>(bufA, twT[3], tb[3], bufB, bufC, CC, 2);

  // l2norm + projection: lx = (f/||f||) @ proj - 0.5
  k_norm_project<<<CC / 16, blk, 0, stream>>>(state_emb, projT, Lxs, CC);
  k_norm_project<<<CC / 16, blk, 0, stream>>>(next_state_emb, projT, Lyn, CC);
  k_norm_project<<<CC / 16, blk, 0, stream>>>(bufC, projT, Lxe, CC);
  k_norm_project<<<VV / 16, blk, 0, stream>>>(term, projT, Lyt, VV);
  k_norm_project<<<1, blk, 0, stream>>>(fxs, projT, Lx0, 1);

  // y-side: exp + column sums S[d]
  k_zero<<<1, 256, 0, stream>>>(Sn, DD);
  k_zero<<<1, 256, 0, stream>>>(St, DD);
  k_exp_colsum<<<(CC * DD) / 4096, 256, 0, stream>>>(Lyn, Wnh, Sn, CC);
  k_exp_colsum<<<(VV * DD) / 4096, 256, 0, stream>>>(Lyt, Wth, St, VV);

  // x-side: row-softmax-normalized U matrices (f16)
  k_make_u<<<CC / 8, blk, 0, stream>>>(Lxs, Sn, Ush, CC);
  k_make_u<<<CC / 8, blk, 0, stream>>>(Lxe, St, Ueh, CC);
  k_make_u<<<1, blk, 0, stream>>>(Lx0, Sn, Uxh, 1);

  // HMM forward recursion in rank-128 space (T graph-captured step launches)
  k_zero<<<(TT * NN + 255) / 256, 256, 0, stream>>>(Zstep, (long)TT * NN);
  k_init<<<1, blk, 0, stream>>>(Uxh, hb + 2 * NN * DD, hb, Zinit);
  for (int t = 0; t < TT; ++t) {
    float* hprev = hb + ((t + 2) % 3) * NN * DD;
    float* hnext = hb + (t % 3) * NN * DD;
    float* hzero = hb + ((t + 1) % 3) * NN * DD;
    const float* Zp = (t == 0) ? Zinit : (Zstep + (t - 1) * NN);
    k_step<<<CC / 64, blk, 0, stream>>>(t, Ush, Wnh, Ueh, Wth, text, hprev,
                                        hnext, hzero, Zp, Zstep + t * NN);
  }
  k_evidence<<<1, blk, 0, stream>>>(Zstep, mask, (float*)d_out);
}
